// LGCN2_28819230556559
// MI455X (gfx1250) — matrix operand order repaired and checked by reading, not compiled, not run on hardware
//
#include <hip/hip_runtime.h>

typedef __attribute__((ext_vector_type(16))) _Float16 v16h;
typedef __attribute__((ext_vector_type(8)))  _Float16 v8h;
typedef __attribute__((ext_vector_type(8)))  float    v8f;
typedef __attribute__((ext_vector_type(4)))  float    f32x4;

namespace {
constexpr int kN  = 50000;   // nodes
constexpr int kNT = 300000;  // edges
constexpr int kR  = 64;      // relations
constexpr int kRP = 16;      // latent pseudo-relations
constexpr int kE  = 32;      // emb dim
constexpr int kC  = 32;      // classes
constexpr int kLW = 64;      // latent MLP width
constexpr int kEdgesPerWave = 4;
}

__device__ __forceinline__ v8f wmma16(v16h a, v16h b, v8f c) {
  // D = A(16x32 f16) * B(32x16 f16) + C(16x16 f32)
  return __builtin_amdgcn_wmma_f32_16x16x32_f16(false, a, false, b, (short)0, c,
                                                false, false);
}

// Async-stage NFLOATS f32 from global into LDS (16B per op, whole block).
// Uses CDNA5 global_load_async_to_lds_b128 (ASYNCcnt) when on gfx1250.
template <int NFLOATS>
__device__ __forceinline__ void stage_async(const float* __restrict__ g,
                                            float* lds, int tid) {
  constexpr int chunks = NFLOATS / 4;  // 16-byte chunks
  for (int c = tid; c < chunks; c += 256) {
#if defined(__gfx1250__)
    const uint32_t laddr = (uint32_t)(uintptr_t)(lds + c * 4);
    const uint32_t goff  = (uint32_t)(c * 16);
    asm volatile("global_load_async_to_lds_b128 %0, %1, %2"
                 :: "v"(laddr), "v"(goff), "s"(g)
                 : "memory");
#else
    ((f32x4*)lds)[c] = ((const f32x4*)g)[c];
#endif
  }
}

__device__ __forceinline__ void wait_async_all() {
#if defined(__gfx1250__)
  asm volatile("s_wait_asynccnt 0" ::: "memory");
#endif
}

// ---------------------------------------------------------------------------
// K0: zero scratch regions (vectorized; element count is a multiple of 4)
// ---------------------------------------------------------------------------
__global__ void zero_kernel(float* __restrict__ p, size_t n) {
  f32x4* p4 = (f32x4*)p;
  const size_t n4 = n / 4;
  size_t i = (size_t)blockIdx.x * blockDim.x + threadIdx.x;
  const size_t stride = (size_t)gridDim.x * blockDim.x;
  const f32x4 z = {0.0f, 0.0f, 0.0f, 0.0f};
  for (; i < n4; i += stride) p4[i] = z;
}

// ---------------------------------------------------------------------------
// K1: lat{1,2} = softmax(relu(nhots@Wa + ba)@Wb + bb)  via WMMA f16->f32
// One wave per 16-edge tile. Weights async-staged (f32) then transposed to
// f16 in LDS so every WMMA B operand is one contiguous 32B LDS load.
// ---------------------------------------------------------------------------
__global__ __launch_bounds__(256)
void mlp_lat_kernel(const float* __restrict__ nhots,
                    const float* __restrict__ W1a, const float* __restrict__ b1a,
                    const float* __restrict__ W1b, const float* __restrict__ b1b,
                    const float* __restrict__ W2a, const float* __restrict__ b2a,
                    const float* __restrict__ W2b, const float* __restrict__ b2b,
                    float* __restrict__ lat1, float* __restrict__ lat2) {
  // smem layout:
  //   [0, 40KB):   f32 staging {W1a,W2a,W1b,W2b}  -- dead after transposition;
  //                first 16KB then reused as per-wave H buffers (8 x 16x64 f16)
  //   [40KB,56KB): sWaT: W{1,2}a transposed, f16   (2 x 64x64)
  //   [56KB,60KB): sWbT: W{1,2}b transposed, f16   (2 x 16x64)
  __shared__ __align__(32) char smem[60 * 1024];
  float*    stage = (float*)smem;
  _Float16* sWaT  = (_Float16*)(smem + 40 * 1024);
  _Float16* sWbT  = (_Float16*)(smem + 56 * 1024);

  const int tid = threadIdx.x;

  // ---- async stage raw f32 weights into LDS ----
  stage_async<kR * kLW>(W1a, stage, tid);                       // [0,4096)
  stage_async<kR * kLW>(W2a, stage + 4096, tid);                // [4096,8192)
  stage_async<kLW * kRP>(W1b, stage + 8192, tid);               // [8192,9216)
  stage_async<kLW * kRP>(W2b, stage + 9216, tid);               // [9216,10240)
  wait_async_all();
  __syncthreads();

  // ---- transpose + f16 convert: B-layout friendly (column-contiguous) ----
  for (int i = tid; i < kR * kLW; i += 256) {
    const int row = i >> 6, col = i & 63;
    sWaT[col * 64 + row]        = (_Float16)stage[i];
    sWaT[4096 + col * 64 + row] = (_Float16)stage[4096 + i];
  }
  for (int i = tid; i < kLW * kRP; i += 256) {
    const int row = i >> 4, col = i & 15;
    sWbT[col * 64 + row]        = (_Float16)stage[8192 + i];
    sWbT[1024 + col * 64 + row] = (_Float16)stage[9216 + i];
  }
  __syncthreads();  // staging region now dead -> reuse as sH

  const int wave = tid >> 5;
  const int lane = tid & 31;
  const int ln   = lane & 15;
  const int half = lane >> 4;
  const int tile = blockIdx.x * 8 + wave;
  if (tile >= kNT / 16) return;  // wave-uniform: EXEC stays all-ones

  _Float16* sH = (_Float16*)smem + wave * (16 * kLW);  // 2KB per wave

  // ---- A tiles: X[16x64] f32 row-major -> two 16x32 f16 A chunks.
  // Per lane the A-layout is one contiguous 16-float run: [h*16,+16).
  const f32x4* xv = (const f32x4*)(nhots + ((size_t)tile * 16 + ln) * kR);
  f32x4 x0 = xv[4 * half],      x1 = xv[4 * half + 1];
  f32x4 x2 = xv[4 * half + 2],  x3 = xv[4 * half + 3];
  f32x4 x4 = xv[8 + 4 * half],  x5 = xv[8 + 4 * half + 1];
  f32x4 x6 = xv[8 + 4 * half + 2], x7 = xv[8 + 4 * half + 3];
  v16h a0, a1;
  #pragma unroll
  for (int e = 0; e < 4; ++e) {
    a0[e]      = (_Float16)x0[e];  a0[4 + e]  = (_Float16)x1[e];
    a0[8 + e]  = (_Float16)x2[e];  a0[12 + e] = (_Float16)x3[e];
    a1[e]      = (_Float16)x4[e];  a1[4 + e]  = (_Float16)x5[e];
    a1[8 + e]  = (_Float16)x6[e];  a1[12 + e] = (_Float16)x7[e];
  }

  const float* basA[2] = {b1a, b2a};
  const float* basB[2] = {b1b, b2b};
  float* outs[2] = {lat1, lat2};

  for (int br = 0; br < 2; ++br) {
    const _Float16* WaT = sWaT + br * 4096;
    const _Float16* WbT = sWbT + br * 1024;

    // H = relu(X @ Wa + ba): 4 col tiles, K=64 (2 WMMAs each)
    #pragma unroll
    for (int j = 0; j < 4; ++j) {
      const float bj = basA[br][j * 16 + ln];
      v8f acc;
      #pragma unroll
      for (int v = 0; v < 8; ++v) acc[v] = bj;
      // transposed storage -> one contiguous v16h per B chunk
      const v16h bt0 = *(const v16h*)(WaT + (j * 16 + ln) * 64 + half * 16);
      const v16h bt1 = *(const v16h*)(WaT + (j * 16 + ln) * 64 + 32 + half * 16);
      acc = wmma16(a0, bt0, acc);
      acc = wmma16(a1, bt1, acc);
      #pragma unroll
      for (int v = 0; v < 8; ++v) {  // D-layout: row = v+8*half, col = ln
        const float r = fmaxf(acc[v], 0.0f);
        sH[(v + half * 8) * kLW + j * 16 + ln] = (_Float16)r;
      }
    }

    // Reload H (row-major) in A layout: contiguous 16B runs
    const v8h* hp = (const v8h*)(sH + ln * kLW);
    const v8h q0 = hp[2 * half],     q1 = hp[2 * half + 1];
    const v8h q2 = hp[4 + 2 * half], q3 = hp[5 + 2 * half];
    const v16h ha0 = __builtin_shufflevector(q0, q1, 0, 1, 2, 3, 4, 5, 6, 7,
                                             8, 9, 10, 11, 12, 13, 14, 15);
    const v16h ha1 = __builtin_shufflevector(q2, q3, 0, 1, 2, 3, 4, 5, 6, 7,
                                             8, 9, 10, 11, 12, 13, 14, 15);

    // Z = H @ Wb + bb (16x16), K=64
    v8f z;
    {
      const float bj = basB[br][ln];
      #pragma unroll
      for (int v = 0; v < 8; ++v) z[v] = bj;
    }
    const v16h c0 = *(const v16h*)(WbT + ln * 64 + half * 16);
    const v16h c1 = *(const v16h*)(WbT + ln * 64 + 32 + half * 16);
    z = wmma16(ha0, c0, z);
    z = wmma16(ha1, c1, z);

    // Row softmax over 16 cols (xor masks < 16 stay inside each lane-half)
    float* lat = outs[br];
    #pragma unroll
    for (int v = 0; v < 8; ++v) {
      float x = z[v];
      float mx = x;
      #pragma unroll
      for (int off = 1; off < 16; off <<= 1) mx = fmaxf(mx, __shfl_xor(mx, off, 32));
      const float e = __expf(x - mx);
      float sum = e;
      #pragma unroll
      for (int off = 1; off < 16; off <<= 1) sum += __shfl_xor(sum, off, 32);
      lat[((size_t)tile * 16 + v + half * 8) * kRP + ln] = e / sum;
    }
  }
}

// ---------------------------------------------------------------------------
// K2: colsum[o*k] += lat1[t,k]; rowsum[s*k] += lat2[t,k]
// k==0 (every edge hits index 0) accumulated in LDS, one atomic per block.
// ---------------------------------------------------------------------------
__global__ __launch_bounds__(256)
void sums_kernel(const int* __restrict__ s_idx, const int* __restrict__ o_idx,
                 const float* __restrict__ lat1, const float* __restrict__ lat2,
                 float* __restrict__ colsum, float* __restrict__ rowsum) {
  __shared__ float sc, sr;
  if (threadIdx.x == 0) { sc = 0.0f; sr = 0.0f; }
  __syncthreads();
  const int t = blockIdx.x * 256 + threadIdx.x;
  float c0 = 0.0f, r0 = 0.0f;
  if (t < kNT) {
    const size_t s = (size_t)s_idx[t];
    const size_t o = (size_t)o_idx[t];
    c0 = lat1[(size_t)t * kRP];
    r0 = lat2[(size_t)t * kRP];
    #pragma unroll
    for (int k = 1; k < kRP; ++k) {
      atomicAdd(&colsum[o * (size_t)k], lat1[(size_t)t * kRP + k]);
      atomicAdd(&rowsum[s * (size_t)k], lat2[(size_t)t * kRP + k]);
    }
  }
  atomicAdd(&sc, c0);  // ds_add_f32
  atomicAdd(&sr, r0);
  __syncthreads();
  if (threadIdx.x == 0) {
    atomicAdd(&colsum[0], sc);
    atomicAdd(&rowsum[0], sr);
  }
}

// ---------------------------------------------------------------------------
// K3: h[s,:] += sum_k (lat1[t,k]/colsum[o*k]) * weights1[(o*k),:]
// One wave per 4 edges; lane == embedding index. Gather rows prefetched.
// ---------------------------------------------------------------------------
__global__ __launch_bounds__(256)
void spmm1_kernel(const int* __restrict__ s_idx, const int* __restrict__ o_idx,
                  const float* __restrict__ lat1, const float* __restrict__ colsum,
                  const float* __restrict__ weights1, float* __restrict__ h) {
  const int w = (blockIdx.x * 256 + threadIdx.x) >> 5;
  const int lane = threadIdx.x & 31;
  const int t0 = w * kEdgesPerWave;
  if (t0 >= kNT) return;

  #pragma unroll
  for (int i = 0; i < kEdgesPerWave; ++i) {  // prefetch all gather rows
    const size_t o = (size_t)o_idx[t0 + i];
    if (lane < kRP)
      __builtin_prefetch(&weights1[(o * (size_t)lane) * kE], 0, 0);
  }
  #pragma unroll
  for (int i = 0; i < kEdgesPerWave; ++i) {
    const int t = t0 + i;
    const size_t s = (size_t)s_idx[t];
    const size_t o = (size_t)o_idx[t];
    float vk = 0.0f;
    if (lane < kRP)
      vk = lat1[(size_t)t * kRP + lane] / colsum[o * (size_t)lane];
    float acc = 0.0f;
    #pragma unroll
    for (int k = 0; k < kRP; ++k) {
      const float v = __shfl(vk, k, 32);
      acc += v * weights1[(o * (size_t)k) * kE + lane];
    }
    atomicAdd(&h[s * kE + lane], acc);
  }
}

// ---------------------------------------------------------------------------
// K4: h = relu(h + bias1)
// ---------------------------------------------------------------------------
__global__ void bias_relu_kernel(float* __restrict__ h, const float* __restrict__ bias1) {
  const int i = blockIdx.x * 256 + threadIdx.x;
  if (i < kN * kE) h[i] = fmaxf(h[i] + bias1[i & (kE - 1)], 0.0f);
}

// ---------------------------------------------------------------------------
// K5: h2[s*k,:] += (lat2[t,k]/rowsum[s*k]) * h[o,:]
// One wave per 4 edges; k==0 row accumulated in LDS per block.
// ---------------------------------------------------------------------------
__global__ __launch_bounds__(256)
void spmm2_kernel(const int* __restrict__ s_idx, const int* __restrict__ o_idx,
                  const float* __restrict__ lat2, const float* __restrict__ rowsum,
                  const float* __restrict__ h, float* __restrict__ h2) {
  __shared__ float acc0[kE];
  if (threadIdx.x < kE) acc0[threadIdx.x] = 0.0f;
  __syncthreads();
  const int w = (blockIdx.x * 256 + threadIdx.x) >> 5;
  const int lane = threadIdx.x & 31;
  const int t0 = w * kEdgesPerWave;
  if (t0 < kNT) {
    #pragma unroll
    for (int i = 0; i < kEdgesPerWave; ++i) {  // prefetch gathered h rows
      const size_t o = (size_t)o_idx[t0 + i];
      __builtin_prefetch(&h[o * kE + lane], 0, 0);
    }
    #pragma unroll
    for (int i = 0; i < kEdgesPerWave; ++i) {
      const int t = t0 + i;
      const size_t s = (size_t)s_idx[t];
      const size_t o = (size_t)o_idx[t];
      const float hv = h[o * kE + lane];
      float vk = 0.0f;
      if (lane < kRP)
        vk = lat2[(size_t)t * kRP + lane] / rowsum[s * (size_t)lane];
      atomicAdd(&acc0[lane], __shfl(vk, 0, 32) * hv);  // k==0 -> LDS
      #pragma unroll
      for (int k = 1; k < kRP; ++k) {
        const float v = __shfl(vk, k, 32);
        atomicAdd(&h2[(s * (size_t)k) * kE + lane], v * hv);
      }
    }
  }
  __syncthreads();
  if (threadIdx.x < kE) atomicAdd(&h2[threadIdx.x], acc0[threadIdx.x]);
}

// ---------------------------------------------------------------------------
// K6: out[n,c] = sum_r h2[r,n,:] @ W2[r,:,c] + bias2
// WMMA; W2 staged transposed in LDS (contiguous B operands); h2 A tiles are
// contiguous float4 runs. One wave per 16-row tile of out.
// ---------------------------------------------------------------------------
__global__ __launch_bounds__(256)
void out_gemm_kernel(const float* __restrict__ h2, const float* __restrict__ weights2,
                     const float* __restrict__ bias2, float* __restrict__ out) {
  __shared__ _Float16 sW2T[kRP * kE * kC];  // per r: [c(32)][k(32)], 32 KB
  for (int i = threadIdx.x; i < kRP * kE * kC; i += 256) {
    const int r = i >> 10, rem = i & 1023, hrow = rem >> 5, c = rem & 31;
    sW2T[(r * kC + c) * kE + hrow] = (_Float16)weights2[i];
  }
  __syncthreads();

  const int wave = threadIdx.x >> 5;
  const int lane = threadIdx.x & 31;
  const int ln   = lane & 15;
  const int half = lane >> 4;
  const int tile = blockIdx.x * 8 + wave;
  if (tile >= kN / 16) return;

  v8f acc[2];
  #pragma unroll
  for (int j = 0; j < 2; ++j) {
    const float bj = bias2[j * 16 + ln];
    #pragma unroll
    for (int v = 0; v < 8; ++v) acc[j][v] = bj;
  }

  for (int r = 0; r < kRP; ++r) {
    const f32x4* hv = (const f32x4*)(h2 + ((size_t)r * kN + (size_t)tile * 16 + ln) * kE);
    if (r + 1 < kRP)
      __builtin_prefetch(h2 + ((size_t)(r + 1) * kN + (size_t)tile * 16 + ln) * kE, 0, 0);
    const f32x4 y0 = hv[4 * half],     y1 = hv[4 * half + 1];
    const f32x4 y2 = hv[4 * half + 2], y3 = hv[4 * half + 3];
    v16h a;
    #pragma unroll
    for (int e = 0; e < 4; ++e) {
      a[e]     = (_Float16)y0[e];  a[4 + e]  = (_Float16)y1[e];
      a[8 + e] = (_Float16)y2[e];  a[12 + e] = (_Float16)y3[e];
    }
    #pragma unroll
    for (int j = 0; j < 2; ++j) {
      const v16h b = *(const v16h*)(sW2T + (r * kC + j * 16 + ln) * kE + half * 16);
      acc[j] = wmma16(a, b, acc[j]);
    }
  }
  #pragma unroll
  for (int j = 0; j < 2; ++j)
    #pragma unroll
    for (int v = 0; v < 8; ++v)
      out[((size_t)tile * 16 + v + half * 8) * kC + j * 16 + ln] = acc[j][v];
}

// ---------------------------------------------------------------------------
extern "C" void kernel_launch(void* const* d_in, const int* in_sizes, int n_in,
                              void* d_out, int out_size, void* d_ws, size_t ws_size,
                              hipStream_t stream) {
  (void)in_sizes; (void)n_in; (void)out_size; (void)ws_size;
  const int*   s_idx    = (const int*)d_in[0];
  const int*   o_idx    = (const int*)d_in[1];
  const float* nhots    = (const float*)d_in[2];
  const float* W1a      = (const float*)d_in[3];
  const float* b1a      = (const float*)d_in[4];
  const float* W1b      = (const float*)d_in[5];
  const float* b1b      = (const float*)d_in[6];
  const float* W2a      = (const float*)d_in[7];
  const float* b2a      = (const float*)d_in[8];
  const float* W2b      = (const float*)d_in[9];
  const float* b2b      = (const float*)d_in[10];
  const float* weights1 = (const float*)d_in[11];
  const float* weights2 = (const float*)d_in[12];
  const float* bias1    = (const float*)d_in[13];
  const float* bias2    = (const float*)d_in[14];
  float* out = (float*)d_out;

  // Workspace (floats): lat1 | lat2 | colsum | rowsum | h | h2  (~153.6 MB)
  float* ws     = (float*)d_ws;
  float* lat1   = ws;
  float* lat2   = lat1 + (size_t)kNT * kRP;
  float* colsum = lat2 + (size_t)kNT * kRP;
  float* rowsum = colsum + (size_t)kN * kRP;
  float* h      = rowsum + (size_t)kN * kRP;
  float* h2     = h + (size_t)kN * kE;
  const size_t zero_elems =
      (size_t)kN * kRP * 2 + (size_t)kN * kE + (size_t)kN * kRP * kE;

  const int spmm_blocks = (kNT / kEdgesPerWave + 7) / 8;  // 8 waves per block

  zero_kernel<<<2048, 256, 0, stream>>>(colsum, zero_elems);
  mlp_lat_kernel<<<(kNT / 16 + 7) / 8, 256, 0, stream>>>(
      nhots, W1a, b1a, W1b, b1b, W2a, b2a, W2b, b2b, lat1, lat2);
  sums_kernel<<<(kNT + 255) / 256, 256, 0, stream>>>(s_idx, o_idx, lat1, lat2,
                                                     colsum, rowsum);
  spmm1_kernel<<<spmm_blocks, 256, 0, stream>>>(s_idx, o_idx, lat1, colsum,
                                                weights1, h);
  bias_relu_kernel<<<(kN * kE + 255) / 256, 256, 0, stream>>>(h, bias1);
  spmm2_kernel<<<spmm_blocks, 256, 0, stream>>>(s_idx, o_idx, lat2, rowsum, h, h2);
  out_gemm_kernel<<<(kN / 16 + 7) / 8, 256, 0, stream>>>(h2, weights2, bias2, out);
}
